// SelfAttention_13314398618250
// MI455X (gfx1250) — matrix operand order repaired and checked by reading, not compiled, not run on hardware
//
#include <hip/hip_runtime.h>

typedef __attribute__((ext_vector_type(16))) _Float16 v16h;
typedef __attribute__((ext_vector_type(8)))  _Float16 v8h;
typedef __attribute__((ext_vector_type(8)))  float    v8f;
typedef __attribute__((ext_vector_type(4)))  float    v4f;

#define LDT 72   // padded LDS row stride in f16 elements (144 B, 16B-aligned)

// ---- fragment loaders matching gfx1250 WMMA 16-bit VGPR layouts ----
__device__ __forceinline__ v16h frag16(const _Float16* p0, const _Float16* p1) {
  v8h lo = *(const v8h*)p0;      // ds_load_b128
  v8h hi = *(const v8h*)p1;      // ds_load_b128
  v16h r;
#pragma unroll
  for (int i = 0; i < 8; ++i) { r[i] = lo[i]; r[i + 8] = hi[i]; }
  return r;
}
// A-matrix 16x32 f16: lane row = l&15; VGPR0-3: k = hf*8 + 0..7, VGPR4-7: k = 16 + hf*8 + 0..7
__device__ __forceinline__ v16h fragA(const _Float16* row, int kk, int hf) {
  const _Float16* p = row + kk + hf * 8;
  return frag16(p, p + 16);
}
// B-matrix 32x16 f16: lane col = l&15; VGPR0-3: k = hf*8 + 0..7, VGPR4-7: k = 16 + hf*8 + 0..7 (staged K-contiguous per column)
__device__ __forceinline__ v16h fragB(const _Float16* row, int kk, int hf) {
  const _Float16* p = row + kk + hf * 8;
  return frag16(p, p + 16);
}

__device__ __forceinline__ void cvt8(const float* src, _Float16* dst) {
  v4f a = *(const v4f*)src;
  v4f b = *(const v4f*)(src + 4);
  v8h h;
#pragma unroll
  for (int i = 0; i < 4; ++i) { h[i] = (_Float16)a[i]; h[i + 4] = (_Float16)b[i]; }
  *(v8h*)dst = h;
}

#define WMMA_F16(A, B, C) \
  __builtin_amdgcn_wmma_f32_16x16x32_f16(false, (A), false, (B), (short)0, (C), false, false)

// ============================================================
// Y[M,N] = X[M,K] * W[N,K]^T + bias[N]   (both projections)
// block = 128 threads = 4 waves; block tile 128x64; wave tile 32x64
// ============================================================
template <typename TIN, typename TOUT>
__global__ __launch_bounds__(128)
void gemm_bias_wmma(const TIN* __restrict__ X, const float* __restrict__ W,
                    const float* __restrict__ bias, TOUT* __restrict__ Y,
                    int M, int N, int K) {
  __shared__ _Float16 As[128 * LDT];
  __shared__ _Float16 Bs[64 * LDT];
  const int tid = threadIdx.x;
  const int lane = tid & 31;
  const int wv = tid >> 5;
  const int hf = lane >> 4;
  const int ln = lane & 15;
  const int m0 = blockIdx.y * 128;
  const int n0 = blockIdx.x * 64;
  const int lrow = tid >> 1;          // 0..63 B staging row
  const int lseg = (tid & 1) * 32;    // 0 or 32 B staging k-segment

  float bsub[4];
#pragma unroll
  for (int s = 0; s < 4; ++s) bsub[s] = bias[n0 + s * 16 + ln];

  v8f acc[2][4] = {{v8f{}, v8f{}, v8f{}, v8f{}}, {v8f{}, v8f{}, v8f{}, v8f{}}};

  for (int k0 = 0; k0 < K; k0 += 64) {
    __syncthreads();
    {  // stage X tile: one full 64-wide row per thread (convert to f16 if needed)
      const TIN* src = X + (size_t)(m0 + tid) * K + k0;
      _Float16* dst = &As[tid * LDT];
#pragma unroll
      for (int i = 0; i < 64; i += 8) {
        if constexpr (sizeof(TIN) == 4) cvt8((const float*)(src + i), dst + i);
        else                            *(v8h*)(dst + i) = *(const v8h*)(src + i);
      }
    }
    {  // stage W tile (rows = output cols, K contiguous — matches W layout)
      const float* src = W + (size_t)(n0 + lrow) * K + k0 + lseg;
      _Float16* dst = &Bs[lrow * LDT + lseg];
#pragma unroll
      for (int i = 0; i < 32; i += 8) cvt8(src + i, dst + i);
    }
    if (k0 + 64 < K) {
      __builtin_prefetch(X + (size_t)(m0 + tid) * K + k0 + 64, 0, 0);
      __builtin_prefetch(W + (size_t)(n0 + lrow) * K + k0 + 64 + lseg, 0, 0);
    }
    __syncthreads();
#pragma unroll
    for (int kk = 0; kk < 64; kk += 32) {
      v16h a0 = fragA(&As[(wv * 32 + ln) * LDT], kk, hf);
      v16h a1 = fragA(&As[(wv * 32 + 16 + ln) * LDT], kk, hf);
#pragma unroll
      for (int s = 0; s < 4; ++s) {
        v16h bf = fragB(&Bs[(s * 16 + ln) * LDT], kk, hf);
        acc[0][s] = WMMA_F16(a0, bf, acc[0][s]);   // B fragment reused 2x
        acc[1][s] = WMMA_F16(a1, bf, acc[1][s]);
      }
    }
  }

  // C/D layout: lane col n = l&15; VGPR r -> row m = hf*8 + r
#pragma unroll
  for (int st = 0; st < 2; ++st) {
#pragma unroll
    for (int s = 0; s < 4; ++s) {
      const int n = n0 + s * 16 + ln;
#pragma unroll
      for (int r = 0; r < 8; ++r) {
        const int m = m0 + wv * 32 + st * 16 + hf * 8 + r;
        Y[(size_t)m * N + n] = (TOUT)(acc[st][s][r] + bsub[s]);
      }
    }
  }
}

// ============================================================
// Flash attention, causal. qkv: [B,T,3072] f16 (q|k|v chunks of 1024).
// block = 128 threads = 4 waves, per (b, head, 128-row q tile).
// Each wave owns 32 q rows (2 strips); KV tiles of 64, online softmax.
// ============================================================
__global__ __launch_bounds__(128)
void attn_flash_wmma(const _Float16* __restrict__ qkv, _Float16* __restrict__ out,
                     int T) {
  __shared__ _Float16 Qs[128 * LDT];
  __shared__ _Float16 Ks[64 * LDT];
  __shared__ _Float16 Vt[64 * LDT];        // V transposed: [d][kv]
  __shared__ _Float16 Ps[4][32 * LDT];     // per-wave P tile 32x64
  const int C = 1024, C3 = 3072, DH = 64;
  const int tid = threadIdx.x;
  const int lane = tid & 31;
  const int wv = tid >> 5;
  const int hf = lane >> 4;
  const int ln = lane & 15;
  const int qt = blockIdx.x;
  const int h  = blockIdx.y;
  const int b  = blockIdx.z;
  const int q0 = qt * 128;
  const size_t offQ = (size_t)b * T * C3 + (size_t)h * DH;
  const size_t offK = offQ + C;
  const size_t offV = offQ + 2 * C;
  const int lrow = tid >> 1;
  const int lseg = (tid & 1) * 32;

  {  // stage Q tile: one full row per thread (f16 copy)
    const _Float16* src = qkv + offQ + (size_t)(q0 + tid) * C3;
    _Float16* dst = &Qs[tid * LDT];
#pragma unroll
    for (int i = 0; i < 64; i += 8) *(v8h*)(dst + i) = *(const v8h*)(src + i);
  }
  __syncthreads();

  // Q fragments live in registers for the whole KV loop: [strip][kk]
  v16h qf[2][2];
#pragma unroll
  for (int st = 0; st < 2; ++st)
#pragma unroll
    for (int kk = 0; kk < 2; ++kk)
      qf[st][kk] = fragA(&Qs[(wv * 32 + st * 16 + ln) * LDT], kk * 32, hf);

  v8f o[2][4] = {{v8f{}, v8f{}, v8f{}, v8f{}}, {v8f{}, v8f{}, v8f{}, v8f{}}};
  float mrow[2][8], lsum[2][8];
#pragma unroll
  for (int st = 0; st < 2; ++st)
#pragma unroll
    for (int r = 0; r < 8; ++r) { mrow[st][r] = -1e30f; lsum[st][r] = 0.f; }

  const int nkt = 2 * qt + 2;          // KV tiles covering cols 0 .. q0+127
  for (int kt = 0; kt < nkt; ++kt) {
    __syncthreads();                   // everyone done reading Ks/Vt
    const int kv0 = kt * 64;
    {  // stage K tile row-major [kv][d]
      const _Float16* src = qkv + offK + (size_t)(kv0 + lrow) * C3 + lseg;
      _Float16* dst = &Ks[lrow * LDT + lseg];
#pragma unroll
      for (int i = 0; i < 32; i += 8) *(v8h*)(dst + i) = *(const v8h*)(src + i);
    }
    {  // stage V transposed [d][kv]
      const _Float16* src = qkv + offV + (size_t)(kv0 + lrow) * C3 + lseg;
#pragma unroll
      for (int i = 0; i < 32; ++i) Vt[(lseg + i) * LDT + lrow] = src[i];
    }
    if (kt + 1 < nkt) {
      __builtin_prefetch(qkv + offK + (size_t)(kv0 + 64 + lrow) * C3 + lseg, 0, 0);
      __builtin_prefetch(qkv + offV + (size_t)(kv0 + 64 + lrow) * C3 + lseg, 0, 0);
    }
    __syncthreads();

    const bool diag = (kv0 + 64 > q0);   // block-uniform: masking only near diagonal

    // per strip: S = Q K^T, mask/scale, online softmax, P -> LDS, rescale O
#pragma unroll
    for (int st = 0; st < 2; ++st) {
      v8f s[4] = {v8f{}, v8f{}, v8f{}, v8f{}};
#pragma unroll
      for (int kk = 0; kk < 2; ++kk)
#pragma unroll
        for (int t = 0; t < 4; ++t) {
          v16h kf = fragB(&Ks[(t * 16 + ln) * LDT], kk * 32, hf);
          s[t] = WMMA_F16(qf[st][kk], kf, s[t]);
        }

      // scale 1/sqrt(64) (+ causal mask on diagonal tiles)
#pragma unroll
      for (int t = 0; t < 4; ++t) {
        const int kc = kv0 + t * 16 + ln;
#pragma unroll
        for (int r = 0; r < 8; ++r) {
          float v = s[t][r] * 0.125f;
          if (diag) {
            const int qr = q0 + wv * 32 + st * 16 + hf * 8 + r;
            v = (kc > qr) ? -1e30f : v;
          }
          s[t][r] = v;
        }
      }

      // online softmax: row stats via 16-lane butterfly (stays within wave32 half)
#pragma unroll
      for (int r = 0; r < 8; ++r) {
        float t0 = fmaxf(fmaxf(s[0][r], s[1][r]), fmaxf(s[2][r], s[3][r]));
#pragma unroll
        for (int xm = 8; xm >= 1; xm >>= 1) t0 = fmaxf(t0, __shfl_xor(t0, xm, 32));
        const float mnew = fmaxf(mrow[st][r], t0);
        float psum = 0.f;
#pragma unroll
        for (int t = 0; t < 4; ++t) {
          const float e = __expf(s[t][r] - mnew);
          psum += e;
          Ps[wv][(st * 16 + hf * 8 + r) * LDT + t * 16 + ln] = (_Float16)e;
        }
#pragma unroll
        for (int xm = 8; xm >= 1; xm >>= 1) psum += __shfl_xor(psum, xm, 32);
        const float alpha = __expf(mrow[st][r] - mnew);
        lsum[st][r] = lsum[st][r] * alpha + psum;
        mrow[st][r] = mnew;
#pragma unroll
        for (int t = 0; t < 4; ++t) o[st][t][r] *= alpha;
      }
    }

    // O += P @ V  (Ps is per-wave; LDS ops in-order within wave; V frags reused 2x)
#pragma unroll
    for (int kk = 0; kk < 2; ++kk) {
      v16h pf0 = fragA(&Ps[wv][(0 * 16 + ln) * LDT], kk * 32, hf);
      v16h pf1 = fragA(&Ps[wv][(1 * 16 + ln) * LDT], kk * 32, hf);
#pragma unroll
      for (int t = 0; t < 4; ++t) {
        v16h vf = fragB(&Vt[(t * 16 + ln) * LDT], kk * 32, hf);
        o[0][t] = WMMA_F16(pf0, vf, o[0][t]);
        o[1][t] = WMMA_F16(pf1, vf, o[1][t]);
      }
    }
  }

  // normalize and write attn output [B,T,1024] f16
#pragma unroll
  for (int st = 0; st < 2; ++st)
#pragma unroll
    for (int r = 0; r < 8; ++r) {
      const float inv = 1.0f / lsum[st][r];
      const size_t m = (size_t)b * T + q0 + wv * 32 + st * 16 + hf * 8 + r;
#pragma unroll
      for (int t = 0; t < 4; ++t)
        out[m * C + h * DH + t * 16 + ln] = (_Float16)(o[st][t][r] * inv);
    }
}

// ============================================================
extern "C" void kernel_launch(void* const* d_in, const int* in_sizes, int n_in,
                              void* d_out, int out_size, void* d_ws, size_t ws_size,
                              hipStream_t stream) {
  (void)in_sizes; (void)n_in; (void)out_size; (void)ws_size;
  const float* x     = (const float*)d_in[0];  // [B,T,1024]
  const float* W_in  = (const float*)d_in[1];  // [3072,1024]
  const float* b_in  = (const float*)d_in[2];  // [3072]
  const float* W_out = (const float*)d_in[3];  // [1024,1024]
  const float* b_out = (const float*)d_in[4];  // [1024]
  float* out = (float*)d_out;

  const int B = 4, T = 2048, C = 1024;
  const int M = B * T;

  _Float16* qkv  = (_Float16*)d_ws;                       // 48 MiB
  _Float16* attn = qkv + (size_t)M * 3 * C;               // +16 MiB

  dim3 blk(128);
  // 1) qkv = x @ W_in^T + b_in   (f32 in, f16 out)
  gemm_bias_wmma<float, _Float16>
      <<<dim3((3 * C) / 64, M / 128), blk, 0, stream>>>(x, W_in, b_in, qkv, M, 3 * C, C);
  // 2) flash attention per (b, h, 128-row q-tile)
  attn_flash_wmma<<<dim3(T / 128, 16, B), blk, 0, stream>>>(qkv, attn, T);
  // 3) out = attn @ W_out^T + b_out  (f16 in, f32 out)
  gemm_bias_wmma<_Float16, float>
      <<<dim3(C / 64, M / 128), blk, 0, stream>>>(attn, W_out, b_out, out, M, C, C);
}
